// MultiLevelEncoder_31877247271216
// MI455X (gfx1250) — matrix operand order repaired and checked by reading, not compiled, not run on hardware
//
#include <hip/hip_runtime.h>
#include <math.h>

typedef __attribute__((ext_vector_type(16))) _Float16     v16h;
typedef __attribute__((ext_vector_type(2)))  __fp16       pk2;   // return type of cvt_pkrtz
typedef __attribute__((ext_vector_type(8)))  float        v8f;
typedef __attribute__((ext_vector_type(8)))  unsigned int v8u;

constexpr int B   = 64;
constexpr int NR  = 50;
constexpr int NG  = 49;
constexpr int NA  = NR + NG;   // 99
constexpr int Dm  = 512;
constexpr int H   = 8;
constexpr int DFF = 2048;
constexpr int NL  = 3;
constexpr int DPE = 32;

constexpr int NP   = 64;    // padded query/token rows (region 50, grid 49 -> 64)
constexpr int NKP  = 112;   // padded key rows for cross concat (99 -> 112)
constexpr int ATTK = 128;   // padded attention row stride

// pack two f32 -> one dword holding (lo,hi) f16 pair, single v_cvt_pk_rz_f16_f32
__device__ __forceinline__ unsigned int pkh(float a, float b){
  pk2 r = __builtin_amdgcn_cvt_pkrtz(a, b);
  return __builtin_bit_cast(unsigned int, r);
}

// ------------------------------------------------------------------
// elementwise helpers
// ------------------------------------------------------------------
__global__ void k_fill0(float* p, int n){
  int i = blockIdx.x*blockDim.x + threadIdx.x; if (i < n) p[i] = 0.f;
}

__global__ void k_pad_copy(float* dst, const float* src, int nTok){
  int i = blockIdx.x*blockDim.x + threadIdx.x;
  int n = B*NP*Dm; if (i >= n) return;
  int d = i % Dm; int r = (i/Dm) % NP; int b = i/(Dm*NP);
  dst[i] = (r < nTok) ? src[((size_t)b*nTok + r)*Dm + d] : 0.f;
}

__global__ void k_add(float* dst, const float* a, const float* b, int n){
  int i = blockIdx.x*blockDim.x + threadIdx.x; if (i < n) dst[i] = a[i] + b[i];
}

__global__ void k_concat_kin(float* kin, const float* xr, const float* xg,
                             const float* posr, const float* posg){
  int i = blockIdx.x*blockDim.x + threadIdx.x;
  int n = B*NKP*Dm; if (i >= n) return;
  int d = i % Dm; int r = (i/Dm) % NKP; int b = i/(Dm*NKP);
  float v = 0.f;
  if (r < NR)      { size_t s = ((size_t)b*NP + r)*Dm + d;      v = xr[s] + posr[s]; }
  else if (r < NA) { size_t s = ((size_t)b*NP + (r-NR))*Dm + d; v = xg[s] + posg[s]; }
  kin[i] = v;
}

__global__ void k_mask(float* mask, const float* src, int nTok){
  int i = blockIdx.x*blockDim.x + threadIdx.x;
  if (i >= B*NP) return;
  int t = i % NP; int b = i / NP;
  float m = 0.f;
  if (t < nTok){
    int cnt = 0;
    const float* p = src + ((size_t)b*nTok + t)*Dm;
    for (int d = 0; d < Dm; ++d) cnt += (p[d] == 0.f);
    m = (cnt != 0) ? 1.f : 0.f;
  }
  mask[i] = m;
}

// ------------------------------------------------------------------
// geometry bias: geo_w[b,h,i,j] = relu( geo(b,i,j,0:128) @ WG_w + WG_b )
// ------------------------------------------------------------------
__global__ void k_geo(float* geow, const float* boxes, const float* wgw, const float* wgb){
  int i = blockIdx.x*blockDim.x + threadIdx.x;
  int n = B*NA*NA; if (i >= n) return;
  int jj = i % NA; int ii = (i/NA) % NA; int b = i/(NA*NA);
  const float* bi = boxes + ((size_t)b*NA + ii)*4;
  const float* bj = boxes + ((size_t)b*NA + jj)*4;
  float cxi = 0.5f*(bi[0]+bi[2]), cyi = 0.5f*(bi[1]+bi[3]);
  float wi  = bi[2]-bi[0]+1.f,    hi  = bi[3]-bi[1]+1.f;
  float cxj = 0.5f*(bj[0]+bj[2]), cyj = 0.5f*(bj[1]+bj[3]);
  float wj  = bj[2]-bj[0]+1.f,    hj  = bj[3]-bj[1]+1.f;
  float pos[4];
  pos[0] = logf(fmaxf(fabsf((cxi-cxj)/wi), 1e-3f));
  pos[1] = logf(fmaxf(fabsf((cyi-cyj)/hi), 1e-3f));
  pos[2] = logf(wi/wj);
  pos[3] = logf(hi/hj);
  float acc[H];
  #pragma unroll
  for (int h = 0; h < H; ++h) acc[h] = wgb[h];
  const float l1000 = logf(1000.f);
  for (int c = 0; c < 4; ++c){
    for (int jd = 0; jd < 16; ++jd){
      float dim = expf(-(float)jd * l1000 / 16.f);
      float m = pos[c]*100.f*dim;
      float s = sinf(m), co = cosf(m);
      const float* wsn = wgw + (c*16+jd)*H;
      const float* wcs = wgw + (64 + c*16+jd)*H;
      #pragma unroll
      for (int h = 0; h < H; ++h) acc[h] += s*wsn[h] + co*wcs[h];
    }
  }
  for (int h = 0; h < H; ++h){
    geow[(((size_t)b*H + h)*NKP + ii)*NKP + jj] = fmaxf(acc[h], 0.f);
  }
}

// ------------------------------------------------------------------
// generic GEMM: C[M,N] = opt_relu( A[M,K] @ W[K,N] + bias + resid )
// f32 in memory (weights L2-resident), f32->f16 pair-packed into LDS,
// v_wmma_f32_16x16x32_f16 math, double-buffered LDS (1 barrier/chunk).
// Block 256 thr = 8 waves; tile 32(M) x 128(N); K-step 32.
// Each wave owns a 16x32 strip: 2 WMMAs sharing one A fragment.
// requires K % 32 == 0, N % 128 == 0 (K,N in {512,2048}).
// ------------------------------------------------------------------
#define APITCH 40    // dwords per k-pair row of A  (32 m + pad; 4*40  -> +32 bank shift)
#define BPITCH 132   // dwords per k-pair row of B  (128 n + pad; 8*132 -> +32 bank shift)

__global__ __launch_bounds__(256)
void k_gemm(const float* A, const float* W, const float* bias,
            const float* resid, float* C, int M, int N, int K, int doRelu){
  __shared__ unsigned int Ap[2][16*APITCH];   // [k-pair][m], f16 pair per dword
  __shared__ unsigned int Bp[2][16*BPITCH];   // [k-pair][n], f16 pair per dword
  int tid  = threadIdx.x;
  int lane = tid & 31, wave = tid >> 5;
  int m0 = blockIdx.x * 32;
  int nb = blockIdx.y * 128;
  // staging coords
  int am  = tid >> 3;            // 0..31 : A row in tile
  int akp = (tid & 7) * 2;       // 0..14 : first of two A k-pairs
  int bp  = tid >> 4;            // 0..15 : B k-pair (rows 2bp, 2bp+1)
  int bn0 = (tid & 15) * 8;      // 0..120: B col offset
  // compute coords
  int tm = (wave >> 2) << 4;     // 0 / 16
  int tn = (wave & 3) << 5;      // 0,32,64,96
  bool hiL = lane >= 16; int l16 = lane & 15;
  int aRow = m0 + am;

  float4 ra, rb0, rb1, rb2, rb3;
  auto loadChunk = [&](int kk){
    ra.x = ra.y = ra.z = ra.w = 0.f;
    if (aRow < M) ra = *(const float4*)&A[(size_t)aRow*K + kk + akp*2];
    const float* wr = &W[(size_t)(kk + 2*bp)*N + nb + bn0];
    rb0 = *(const float4*)wr;
    rb1 = *(const float4*)(wr+4);
    rb2 = *(const float4*)(wr+N);
    rb3 = *(const float4*)(wr+N+4);
    if (kk + 32 < K)
      __builtin_prefetch(&W[(size_t)(kk+32+2*bp)*N + nb + bn0], 0, 1);
  };
  auto stage = [&](int buf){
    Ap[buf][(akp  )*APITCH + am] = pkh(ra.x, ra.y);
    Ap[buf][(akp+1)*APITCH + am] = pkh(ra.z, ra.w);
    uint4 u0, u1;
    u0.x = pkh(rb0.x, rb2.x); u0.y = pkh(rb0.y, rb2.y);
    u0.z = pkh(rb0.z, rb2.z); u0.w = pkh(rb0.w, rb2.w);
    u1.x = pkh(rb1.x, rb3.x); u1.y = pkh(rb1.y, rb3.y);
    u1.z = pkh(rb1.z, rb3.z); u1.w = pkh(rb1.w, rb3.w);
    *(uint4*)&Bp[buf][bp*BPITCH + bn0]     = u0;
    *(uint4*)&Bp[buf][bp*BPITCH + bn0 + 4] = u1;
  };

  v8f c0 = {}, c1 = {};
  int nIter = K >> 5;
  loadChunk(0);
  stage(0);
  for (int it = 0; it < nIter; ++it){
    int buf = it & 1;
    bool more = (it + 1 < nIter);
    if (more) loadChunk((it+1) << 5);   // issue next-chunk global loads early
    __syncthreads();                    // staged chunk `it` visible
    v8u au, bu0, bu1;
    #pragma unroll
    for (int i = 0; i < 8; ++i){
      int kpA = ((i < 4) ? i : i + 4) + (hiL ? 4 : 0);
      au[i]  = Ap[buf][kpA*APITCH + tm + l16];
      int kpB = ((i < 4) ? i : i + 4) + (hiL ? 4 : 0);
      bu0[i] = Bp[buf][kpB*BPITCH + tn + l16];
      bu1[i] = Bp[buf][kpB*BPITCH + tn + 16 + l16];
    }
    v16h af  = __builtin_bit_cast(v16h, au);
    v16h bf0 = __builtin_bit_cast(v16h, bu0);
    v16h bf1 = __builtin_bit_cast(v16h, bu1);
    c0 = __builtin_amdgcn_wmma_f32_16x16x32_f16(false, af, false, bf0, (short)0, c0, false, false);
    c1 = __builtin_amdgcn_wmma_f32_16x16x32_f16(false, af, false, bf1, (short)0, c1, false, false);
    if (more) stage(buf ^ 1);           // convert+store next chunk (other buffer)
  }
  // epilogue (C layout: VGPR v -> row tm+v (+8 if lane>=16), col tn+lane%16)
  int colA = nb + tn + l16;
  int colB = colA + 16;
  int rbase = m0 + tm + (hiL ? 8 : 0);
  float bvA = bias ? bias[colA] : 0.f;
  float bvB = bias ? bias[colB] : 0.f;
  #pragma unroll
  for (int v = 0; v < 8; ++v){
    int row = rbase + v;
    if (row < M){
      float vA = c0[v] + bvA;
      float vB = c1[v] + bvB;
      if (resid){
        vA += resid[(size_t)row*N + colA];
        vB += resid[(size_t)row*N + colB];
      }
      if (doRelu){ vA = fmaxf(vA, 0.f); vB = fmaxf(vB, 0.f); }
      C[(size_t)row*N + colA] = vA;
      C[(size_t)row*N + colB] = vB;
    }
  }
}

// ------------------------------------------------------------------
// attention scores: ATT[b,h,q,k] = (Q[b,q,h,:] . K[b,k,h,:]) * scale
// one wave per 16x16 tile; fragments built from global float2 (L2-hit).
// ------------------------------------------------------------------
__global__ __launch_bounds__(256)
void k_scores(const float* Q, const float* K, float* ATT,
              int KROWS, int NkT, float scale){
  int lane = threadIdx.x & 31, wave = threadIdx.x >> 5;
  int bh = blockIdx.x; int b = bh / H, h = bh % H;
  int q0 = blockIdx.y * 16;
  if (wave >= NkT) return;
  int n0 = wave * 16;
  bool hiL = lane >= 16; int l16 = lane & 15;
  v8f c = {};
  #pragma unroll
  for (int ch = 0; ch < 2; ++ch){     // DK=64 -> 2 chunks of 32
    const float* qr = Q + ((size_t)b*NP    + q0 + l16)*Dm + h*64 + ch*32;
    const float* kr = K + ((size_t)b*KROWS + n0 + l16)*Dm + h*64 + ch*32;
    v8u au, bu;
    #pragma unroll
    for (int i = 0; i < 8; ++i){
      int ka = ((i < 4) ? 2*i : 16 + 2*(i-4)) + (hiL ? 8 : 0);
      float2 qa = *(const float2*)&qr[ka];
      au[i] = pkh(qa.x, qa.y);
      int kb = ((i < 4) ? 2*i : 16 + 2*(i-4)) + (hiL ? 8 : 0);
      float2 kv = *(const float2*)&kr[kb];
      bu[i] = pkh(kv.x, kv.y);
    }
    v16h af = __builtin_bit_cast(v16h, au);
    v16h bf = __builtin_bit_cast(v16h, bu);
    c = __builtin_amdgcn_wmma_f32_16x16x32_f16(false, af, false, bf, (short)0, c, false, false);
  }
  int col = n0 + l16;
  int rbase = q0 + (hiL ? 8 : 0);
  #pragma unroll
  for (int v = 0; v < 8; ++v){
    ATT[(((size_t)b*H + h)*NP + rbase + v)*ATTK + col] = c[v]*scale;
  }
}

// ------------------------------------------------------------------
// softmax + mask + log(geo) per row; zero-pads the rest of the row.
// mode 0: region self (maskKey), 1: grid self (maskKey), 2: cross (aligns)
// ------------------------------------------------------------------
__global__ void k_softmax(float* ATT, const float* GEOW, const float* maskKey,
                          const float* aligns, int mode, int nkValid,
                          int qoff, int koff, int nTokQ){
  int i = blockIdx.x*blockDim.x + threadIdx.x;
  if (i >= B*H*NP) return;
  int q = i % NP; int h = (i/NP) % H; int b = i/(NP*H);
  float* row = ATT + (size_t)i*ATTK;
  int qc = min(q, nTokQ-1);
  const float* grow = GEOW + (((size_t)b*H + h)*NKP + qoff + qc)*NKP + koff;
  float mx = -1e30f;
  for (int k = 0; k < nkValid; ++k){
    bool masked;
    if (mode != 2) masked = (maskKey[b*NP + k] != 0.f);
    else {
      if (k < NR) masked = (aligns[((size_t)b*NR + k)*NG + qc] == 0.f);
      else        masked = ((k - NR) != qc);
    }
    float s = masked ? -1e9f : row[k];
    s += logf(fmaxf(grow[k], 1e-6f));
    row[k] = s;
    mx = fmaxf(mx, s);
  }
  float sum = 0.f;
  for (int k = 0; k < nkValid; ++k){ float e = expf(row[k]-mx); row[k] = e; sum += e; }
  float inv = 1.f/sum;
  for (int k = 0; k < nkValid; ++k) row[k] *= inv;
  for (int k = nkValid; k < ATTK; ++k) row[k] = 0.f;
}

// ------------------------------------------------------------------
// attention output: OUT[b,q,h,:] = wgt[b,h,q,:] @ V[b,:,h,:]
// KC = number of 32-wide K chunks (2 self / 4 cross), fully unrolled.
// ------------------------------------------------------------------
template<int KC>
__global__ __launch_bounds__(128)
void k_attout(const float* ATT, const float* V, float* OUT, int VROWS){
  int lane = threadIdx.x & 31, wave = threadIdx.x >> 5;
  int bh = blockIdx.x; int b = bh / H, h = bh % H;
  int q0 = blockIdx.y * 16;
  int d0 = wave * 16;                 // DV=64 -> 4 waves
  bool hiL = lane >= 16; int l16 = lane & 15;
  v8f c = {};
  #pragma unroll
  for (int ch = 0; ch < KC; ++ch){
    const float* ar = ATT + (((size_t)b*H + h)*NP + q0 + l16)*ATTK + ch*32;
    v8u au, bu;
    #pragma unroll
    for (int i = 0; i < 8; ++i){
      int ka = ((i < 4) ? 2*i : 16 + 2*(i-4)) + (hiL ? 8 : 0);
      float2 av = *(const float2*)&ar[ka];
      au[i] = pkh(av.x, av.y);
      int kb = ((i < 4) ? 2*i : 16 + 2*(i-4)) + (hiL ? 8 : 0);
      int kr0 = min(ch*32 + kb,     VROWS-1);   // pad rows clamped; wgt=0 there
      int kr1 = min(ch*32 + kb + 1, VROWS-1);
      bu[i] = pkh(V[((size_t)b*VROWS + kr0)*Dm + h*64 + d0 + l16],
                  V[((size_t)b*VROWS + kr1)*Dm + h*64 + d0 + l16]);
    }
    v16h af = __builtin_bit_cast(v16h, au);
    v16h bf = __builtin_bit_cast(v16h, bu);
    c = __builtin_amdgcn_wmma_f32_16x16x32_f16(false, af, false, bf, (short)0, c, false, false);
  }
  int col = d0 + l16;
  int rbase = q0 + (hiL ? 8 : 0);
  #pragma unroll
  for (int v = 0; v < 8; ++v){
    OUT[((size_t)b*NP + rbase + v)*Dm + h*64 + col] = c[v];
  }
}

// ------------------------------------------------------------------
// layer norm over D=512, block per row; optional fused residual add
// ------------------------------------------------------------------
__global__ __launch_bounds__(256)
void k_ln(float* out, const float* x, const float* resid,
          const float* g, const float* bta, int rows){
  int row = blockIdx.x; if (row >= rows) return;
  int t = threadIdx.x;
  __shared__ float red[256];
  const float* xr = x + (size_t)row*Dm;
  float v0 = xr[t], v1 = xr[t+256];
  if (resid){ const float* rr = resid + (size_t)row*Dm; v0 += rr[t]; v1 += rr[t+256]; }
  red[t] = v0 + v1; __syncthreads();
  for (int s = 128; s > 0; s >>= 1){ if (t < s) red[t] += red[t+s]; __syncthreads(); }
  float mean = red[0] * (1.f/Dm);
  __syncthreads();
  float d0 = v0 - mean, d1 = v1 - mean;
  red[t] = d0*d0 + d1*d1; __syncthreads();
  for (int s = 128; s > 0; s >>= 1){ if (t < s) red[t] += red[t+s]; __syncthreads(); }
  float rstd = rsqrtf(red[0]*(1.f/Dm) + 1e-5f);
  out[(size_t)row*Dm + t]       = d0*rstd*g[t]     + bta[t];
  out[(size_t)row*Dm + t + 256] = d1*rstd*g[t+256] + bta[t+256];
}

// ------------------------------------------------------------------
// GPO pooling weights: bi-GRU over positional encodings + softmax(T=0.1)
// ------------------------------------------------------------------
__device__ __forceinline__ float sigm(float x){ return 1.f/(1.f+expf(-x)); }

__global__ __launch_bounds__(96)
void k_gpo(float* gpow,
           const float* WiF, const float* WhF, const float* biF, const float* bhF,
           const float* WiB, const float* WhB, const float* biB, const float* bhB,
           const float* lin){
  __shared__ float pe[NG][DPE];
  __shared__ float xi[NG][3*DPE];
  __shared__ float hcur[DPE];
  __shared__ float gh[3*DPE];
  __shared__ float outm[NG][DPE];
  int tid = threadIdx.x;     // 96 threads
  for (int e = tid; e < NG*DPE; e += 96){
    int t = e/DPE, d = e%DPE;
    int i2 = d >> 1;
    float div = expf(-(float)(2*i2)*logf(10000.f)/(float)DPE);
    float ang = (float)t*div;
    pe[t][d] = (d & 1) ? cosf(ang) : sinf(ang);
    outm[t][d] = 0.f;
  }
  __syncthreads();
  // forward
  for (int t = 0; t < NG; ++t){
    float s = biF[tid];
    for (int d = 0; d < DPE; ++d) s += pe[t][d]*WiF[tid*DPE + d];
    xi[t][tid] = s;
  }
  if (tid < DPE) hcur[tid] = 0.f;
  __syncthreads();
  for (int t = 0; t < NG; ++t){
    float s = bhF[tid];
    for (int d = 0; d < DPE; ++d) s += hcur[d]*WhF[tid*DPE + d];
    gh[tid] = s;
    __syncthreads();
    float hn = 0.f;
    if (tid < DPE){
      float r  = sigm(xi[t][tid]        + gh[tid]);
      float z  = sigm(xi[t][DPE+tid]    + gh[DPE+tid]);
      float nn = tanhf(xi[t][2*DPE+tid] + r*gh[2*DPE+tid]);
      hn = (1.f-z)*nn + z*hcur[tid];
    }
    __syncthreads();
    if (tid < DPE){ hcur[tid] = hn; outm[t][tid] += 0.5f*hn; }
    __syncthreads();
  }
  // backward (reversed sequence)
  for (int t = 0; t < NG; ++t){
    float s = biB[tid];
    for (int d = 0; d < DPE; ++d) s += pe[NG-1-t][d]*WiB[tid*DPE + d];
    xi[t][tid] = s;
  }
  if (tid < DPE) hcur[tid] = 0.f;
  __syncthreads();
  for (int t = 0; t < NG; ++t){
    float s = bhB[tid];
    for (int d = 0; d < DPE; ++d) s += hcur[d]*WhB[tid*DPE + d];
    gh[tid] = s;
    __syncthreads();
    float hn = 0.f;
    if (tid < DPE){
      float r  = sigm(xi[t][tid]        + gh[tid]);
      float z  = sigm(xi[t][DPE+tid]    + gh[DPE+tid]);
      float nn = tanhf(xi[t][2*DPE+tid] + r*gh[2*DPE+tid]);
      hn = (1.f-z)*nn + z*hcur[tid];
    }
    __syncthreads();
    if (tid < DPE){ hcur[tid] = hn; outm[NG-1-t][tid] += 0.5f*hn; }
    __syncthreads();
  }
  if (tid == 0){
    float sc[NG]; float mx = -1e30f;
    for (int t = 0; t < NG; ++t){
      float s = 0.f;
      for (int d = 0; d < DPE; ++d) s += outm[t][d]*lin[d];
      sc[t] = s*10.f;                 // scores / 0.1
      mx = fmaxf(mx, sc[t]);
    }
    float sum = 0.f;
    for (int t = 0; t < NG; ++t){ sc[t] = expf(sc[t]-mx); sum += sc[t]; }
    for (int t = 0; t < NG; ++t) gpow[t] = sc[t]/sum;
    for (int t = NG; t < NP; ++t) gpow[t] = 0.f;
  }
}

// ------------------------------------------------------------------
// descending sort over tokens + weighted pool accumulate
// ------------------------------------------------------------------
__global__ void k_sortpool(float* feats, const float* XG, const float* gpow){
  int i = blockIdx.x*blockDim.x + threadIdx.x;
  if (i >= B*Dm) return;
  int d = i % Dm; int b = i / Dm;
  float v[NG];
  for (int t = 0; t < NG; ++t) v[t] = XG[((size_t)b*NP + t)*Dm + d];
  for (int a = 1; a < NG; ++a){
    float key = v[a]; int j = a-1;
    while (j >= 0 && v[j] < key){ v[j+1] = v[j]; --j; }
    v[j+1] = key;
  }
  float acc = 0.f;
  for (int t = 0; t < NG; ++t) acc += v[t]*gpow[t];
  feats[i] += acc;
}

// ------------------------------------------------------------------
// final L2 normalize
// ------------------------------------------------------------------
__global__ __launch_bounds__(256)
void k_norm(float* out, const float* feats){
  int b = blockIdx.x; int t = threadIdx.x;
  __shared__ float red[256];
  const float* f = feats + (size_t)b*Dm;
  float v0 = f[t], v1 = f[t+256];
  red[t] = v0*v0 + v1*v1; __syncthreads();
  for (int s = 128; s > 0; s >>= 1){ if (t < s) red[t] += red[t+s]; __syncthreads(); }
  float inv = 1.f/(sqrtf(red[0]) + 1e-8f);
  out[(size_t)b*Dm + t]       = v0*inv;
  out[(size_t)b*Dm + t + 256] = v1*inv;
}

// ------------------------------------------------------------------
// host orchestration
// ------------------------------------------------------------------
extern "C" void kernel_launch(void* const* d_in, const int* in_sizes, int n_in,
                              void* d_out, int out_size, void* d_ws, size_t ws_size,
                              hipStream_t stream){
  auto pf = [&](int i){ return (const float*)d_in[i]; };
  const float* regions = pf(0);
  const float* grids   = pf(1);
  const float* boxes   = pf(2);
  const float* aligns  = pf(3);
  const float* rembed  = pf(4);
  const float* gembed  = pf(5);
  const int PBASE = 6;
  const int WGW   = PBASE + 3*NL*18;   // 168: WG_w, 169: WG_b
  const int GRU   = WGW + 2;           // 170..177: fwd{Wi,Wh,bi,bh}, bwd{...}
  const int LIN   = GRU + 8;           // 178: gpo_lin

  float* ws = (float*)d_ws;
  size_t o = 0;
  auto alloc = [&](size_t n){ float* p = ws + o; o += n; return p; };
  float* XR    = alloc((size_t)B*NP*Dm);
  float* XGb   = alloc((size_t)B*NP*Dm);
  float* POSR  = alloc((size_t)B*NP*Dm);
  float* POSG  = alloc((size_t)B*NP*Dm);
  float* QIN   = alloc((size_t)B*NP*Dm);
  float* KIN   = alloc((size_t)B*NKP*Dm);
  float* QB    = alloc((size_t)B*NP*Dm);
  float* KB    = alloc((size_t)B*NKP*Dm);
  float* VB    = alloc((size_t)B*NKP*Dm);
  float* ATT   = alloc((size_t)B*H*NP*ATTK);
  float* AOUT  = alloc((size_t)B*NP*Dm);
  float* T1    = alloc((size_t)B*NP*Dm);
  float* T2    = alloc((size_t)B*NP*Dm);
  float* FF    = alloc((size_t)B*NP*DFF);
  float* GEOW  = alloc((size_t)B*H*NKP*NKP);
  float* MASKR = alloc((size_t)B*NP);
  float* MASKG = alloc((size_t)B*NP);
  float* GPOW  = alloc((size_t)NP);
  float* FEATS = alloc((size_t)B*Dm);

  auto cdiv = [](int a, int b){ return (a + b - 1)/b; };
  int n1 = B*NP*Dm;

  k_fill0   <<<cdiv(B*Dm,256),256,0,stream>>>(FEATS, B*Dm);
  k_pad_copy<<<cdiv(n1,256),256,0,stream>>>(XR,   regions, NR);
  k_pad_copy<<<cdiv(n1,256),256,0,stream>>>(XGb,  grids,   NG);
  k_pad_copy<<<cdiv(n1,256),256,0,stream>>>(POSR, rembed,  NR);
  k_pad_copy<<<cdiv(n1,256),256,0,stream>>>(POSG, gembed,  NG);
  k_mask    <<<cdiv(B*NP,256),256,0,stream>>>(MASKR, regions, NR);
  k_mask    <<<cdiv(B*NP,256),256,0,stream>>>(MASKG, grids,   NG);
  k_geo     <<<cdiv(B*NA*NA,256),256,0,stream>>>(GEOW, boxes, pf(WGW), pf(WGW+1));
  k_gpo     <<<1,96,0,stream>>>(GPOW, pf(GRU+0),pf(GRU+1),pf(GRU+2),pf(GRU+3),
                                       pf(GRU+4),pf(GRU+5),pf(GRU+6),pf(GRU+7), pf(LIN));

  auto gemm = [&](const float* A, const float* W, const float* bias, const float* resid,
                  float* C, int M, int N, int K, int relu){
    dim3 g(cdiv(M,32), N/128);
    k_gemm<<<g,256,0,stream>>>(A, W, bias, resid, C, M, N, K, relu);
  };
  auto lnf = [&](float* out, const float* x, const float* resid,
                 const float* g, const float* b, int rows){
    k_ln<<<rows,256,0,stream>>>(out, x, resid, g, b, rows);
  };

  const float scale = 0.125f;   // 1/sqrt(DK)

  for (int L = 0; L < NL; ++L){
    for (int grp = 0; grp < 3; ++grp){   // 0=region self, 1=grid self, 2=cross
      int base = PBASE + (grp*NL + L)*18;
      const float *Wq=pf(base+0),  *bq=pf(base+1),  *Wk=pf(base+2),  *bk=pf(base+3),
                  *Wv=pf(base+4),  *bv=pf(base+5),  *Wo=pf(base+6),  *bo=pf(base+7),
                  *lmg=pf(base+8), *lmb=pf(base+9), *lag=pf(base+10),*lab=pf(base+11),
                  *W1=pf(base+12), *b1=pf(base+13), *W2=pf(base+14), *b2=pf(base+15),
                  *lfg=pf(base+16),*lfb=pf(base+17);
      float* X; const float* POS; int nTokQ, nkValid, qoff, koff, mode;
      if (grp == 0){ X = XR;  POS = POSR; nTokQ = NR; nkValid = NR; qoff = 0;  koff = 0;  mode = 0; }
      else if (grp == 1){ X = XGb; POS = POSG; nTokQ = NG; nkValid = NG; qoff = NR; koff = NR; mode = 1; }
      else { X = XGb; POS = POSG; nTokQ = NG; nkValid = NA; qoff = NR; koff = 0;  mode = 2; }

      k_add<<<cdiv(n1,256),256,0,stream>>>(QIN, X, POS, n1);     // q_in = queries + pos_q
      const float* KSRC; int KROWS;
      if (grp == 2){
        k_concat_kin<<<cdiv(B*NKP*Dm,256),256,0,stream>>>(KIN, XR, XGb, POSR, POSG);
        KSRC = KIN; KROWS = NKP;
      } else { KSRC = QIN; KROWS = NP; }

      int MQ = B*NP, MK = B*KROWS;
      gemm(QIN,  Wq, bq, nullptr, QB, MQ, Dm, Dm, 0);
      gemm(KSRC, Wk, bk, nullptr, KB, MK, Dm, Dm, 0);
      gemm(KSRC, Wv, bv, nullptr, VB, MK, Dm, Dm, 0);

      int NkT = cdiv(nkValid, 16);
      k_scores<<<dim3(B*H, NP/16),256,0,stream>>>(QB, KB, ATT, KROWS, NkT, scale);
      k_softmax<<<cdiv(B*H*NP,256),256,0,stream>>>(ATT, GEOW, (mode==1)?MASKG:MASKR,
                                                   aligns, mode, nkValid, qoff, koff, nTokQ);
      if (grp == 2) k_attout<4><<<dim3(B*H, NP/16),128,0,stream>>>(ATT, VB, AOUT, KROWS);
      else          k_attout<2><<<dim3(B*H, NP/16),128,0,stream>>>(ATT, VB, AOUT, KROWS);

      gemm(AOUT, Wo, bo, QIN, T1, MQ, Dm, Dm, 0);   // + residual q_in
      lnf(T2, T1, nullptr, lmg, lmb, MQ);           // ln_mha
      lnf(T1, X,  T2,      lag, lab, MQ);           // ln_att(queries + att)
      gemm(T1, W1, b1, nullptr, FF, MQ, DFF, Dm, 1);          // relu(x@W1+b1)
      gemm(FF, W2, b2, T1,      T2, MQ, Dm, DFF, 0);          // + residual att
      lnf(X, T2, nullptr, lfg, lfb, MQ);            // ln_ff -> new stream state
    }
    k_sortpool<<<cdiv(B*Dm,256),256,0,stream>>>(FEATS, XGb, GPOW);
  }
  k_norm<<<B,256,0,stream>>>((float*)d_out, FEATS);
  (void)in_sizes; (void)n_in; (void)out_size; (void)ws_size;
}
